// Relative_Multihead_Attention_80178449482220
// MI455X (gfx1250) — compile-verified, matrix operand order checked by reading, not run on hardware
//
#include <hip/hip_runtime.h>
#include <hip/hip_bf16.h>

#define B_   2
#define Q_   2048
#define M_   2048
#define H_   16
#define D_   64
#define HID_ 1024
#define K_   4096   // K_LEN = M + Q

// Async global->LDS staging (GLOBAL_LOAD_ASYNC_TO_LDS_B128, ASYNCcnt).
// Flip to 0 if the assembler rejects the mnemonic/operands.
#define USE_ASYNC_LDS 1

typedef __attribute__((ext_vector_type(16))) __bf16 v16bf;
typedef __attribute__((ext_vector_type(8)))  float  v8f;

__device__ __forceinline__ __bf16 f2bf(float f) {
  unsigned u = __builtin_bit_cast(unsigned, f);
  u += 0x7fffu + ((u >> 16) & 1u);            // round-to-nearest-even
  unsigned short h = (unsigned short)(u >> 16);
  return __builtin_bit_cast(__bf16, h);
}

// One 16B chunk global->LDS. Async form: single instruction, no VGPR
// round-trip; LDS address = low 32 bits of the generic pointer (ISA 10.2:
// flat->LDS truncates to addr[31:0]).
__device__ __forceinline__ void g2l_b128(void* lds, const void* gm) {
#if USE_ASYNC_LDS
  asm volatile("global_load_async_to_lds_b128 %0, %1, off"
               :
               : "v"((unsigned)(uintptr_t)lds),
                 "v"((unsigned long long)(uintptr_t)gm)
               : "memory");
#else
  *reinterpret_cast<uint4*>(lds) = *reinterpret_cast<const uint4*>(gm);
#endif
}
__device__ __forceinline__ void async_join() {
#if USE_ASYNC_LDS
  asm volatile("s_wait_asynccnt 0" ::: "memory");
#endif
}

// Universal 16x32 bf16 WMMA fragment load from LDS (row stride ld, elements).
// Lane (r = lane&15, base = lane<16?0:8) holds halves k = base..base+7 and
// base+16..base+23 of row r: two contiguous 16B runs -> two ds_load_b128.
struct U4x2 { uint4 lo, hi; };
__device__ __forceinline__ v16bf load_frag(const __bf16* p, int ld) {
  int lane = threadIdx.x & 31;
  const __bf16* q = p + (lane & 15) * ld + ((lane < 16) ? 0 : 8);
  U4x2 u;
  u.lo = *reinterpret_cast<const uint4*>(q);
  u.hi = *reinterpret_cast<const uint4*>(q + 16);
  return __builtin_bit_cast(v16bf, u);
}

__device__ __forceinline__ v8f wmma_bf16(v16bf a, v16bf b, v8f c) {
  return __builtin_amdgcn_wmma_f32_16x16x32_bf16(false, a, false, b, (short)0, c,
                                                 false, false);
}

__device__ __forceinline__ uint4 pack8_bf(v8f a) {
  struct P8 { __bf16 h[8]; } z;
#pragma unroll
  for (int e = 0; e < 8; ++e) z.h[e] = f2bf(a[e]);
  return __builtin_bit_cast(uint4, z);
}

// ------------------------------------------------------------- transpose cvt
// Wt[n][k] = (bf16) W[k][n], 1024x1024, LDS-tiled so both sides stay coalesced
__global__ __launch_bounds__(256)
void cvt_transpose_kernel(const float* __restrict__ s, __bf16* __restrict__ d) {
  __shared__ float tile[32][33];
  int bx = blockIdx.x, by = blockIdx.y;
  int tx = threadIdx.x & 31, ty = threadIdx.x >> 5;
#pragma unroll
  for (int r = 0; r < 4; ++r)
    tile[ty + 8 * r][tx] = s[(size_t)(by * 32 + ty + 8 * r) * HID_ + bx * 32 + tx];
  __syncthreads();
#pragma unroll
  for (int r = 0; r < 4; ++r)
    d[(size_t)(bx * 32 + ty + 8 * r) * HID_ + by * 32 + tx] =
        f2bf(tile[tx][ty + 8 * r]);
}

// One block (256 thr) per row of 1024; concat mode builds kv = [memory;query]
__global__ __launch_bounds__(256)
void layernorm_kernel(const float* __restrict__ query,
                      const float* __restrict__ memory,
                      const float* __restrict__ gamma,
                      const float* __restrict__ beta,
                      __bf16* __restrict__ out, int concat) {
  int row = blockIdx.x;
  const float* src;
  if (concat) {
    int b = row / K_, t = row % K_;
    src = (t < M_) ? (memory + ((size_t)b * M_ + t) * HID_)
                   : (query + ((size_t)b * Q_ + (t - M_)) * HID_);
  } else {
    src = query + (size_t)row * HID_;
  }
  int tid = threadIdx.x;
  float x[4];
  float s1 = 0.f, s2 = 0.f;
#pragma unroll
  for (int t = 0; t < 4; ++t) {
    x[t] = src[tid + 256 * t];
    s1 += x[t];
    s2 += x[t] * x[t];
  }
  __shared__ float r1[256], r2[256];
  r1[tid] = s1; r2[tid] = s2;
  __syncthreads();
  for (int off = 128; off > 0; off >>= 1) {
    if (tid < off) { r1[tid] += r1[tid + off]; r2[tid] += r2[tid + off]; }
    __syncthreads();
  }
  float mean = r1[0] * (1.0f / HID_);
  float var  = r2[0] * (1.0f / HID_) - mean * mean;
  float rstd = rsqrtf(var + 1e-3f);
  __bf16* o = out + (size_t)row * HID_;
#pragma unroll
  for (int t = 0; t < 4; ++t) {
    int c = tid + 256 * t;
    o[c] = f2bf((x[t] - mean) * rstd * gamma[c] + beta[c]);
  }
}

__global__ void posemb_kernel(__bf16* __restrict__ out) {
  int idx = blockIdx.x * blockDim.x + threadIdx.x;
  if (idx >= K_ * HID_) return;
  int j = idx >> 10, i = idx & 1023;
  float pos = (float)(K_ - 1 - j);
  int f = (i < 512) ? i : (i - 512);
  float invf = __expf(-((float)(2 * f) / (float)HID_) * 9.210340371976184f);
  float a = pos * invf;
  out[idx] = f2bf((i < 512) ? __sinf(a) : __cosf(a));
}

// ------------------------------------------------- WMMA GEMM, 128x128 tiles
// C = A[rowsM,1024] x W[1024,1024], W given pre-transposed (Wt[n][k]).
// MODE 0: f32 flat out.        MODE 1: bf16 head-major out1[h][rowsM][64].
// MODE 2: dual head-major + f32 bias (q +r_w_bias / +r_r_bias).
// MODE 3: bf16 transposed out1[h][64][rowsM] (for V), packed b128 stores.
template <int MODE>
__global__ __launch_bounds__(256)
void gemm_bf16_kernel(const __bf16* __restrict__ A,
                      const __bf16* __restrict__ Wt, int rowsM,
                      float* __restrict__ outF, __bf16* __restrict__ out1,
                      __bf16* __restrict__ out2,
                      const float* __restrict__ bias1,
                      const float* __restrict__ bias2) {
  __shared__ alignas(16) __bf16 As[128 * 64];
  __shared__ alignas(16) __bf16 Bs[128 * 64];   // n-major: Bs[n][k]
  int tid = threadIdx.x;
  int w = tid >> 5, lane = tid & 31;
  int mblk = blockIdx.x, nblk = blockIdx.y;
  int wr = w & 3, wc = w >> 2;            // wave tile: rows 32*wr, cols 64*wc
  v8f acc[2][4];
#pragma unroll
  for (int i = 0; i < 2; ++i)
#pragma unroll
    for (int t = 0; t < 4; ++t) acc[i][t] = (v8f){};

  for (int kb = 0; kb < HID_ / 64; ++kb) {
    for (int c = tid; c < 1024; c += 256) {      // A tile 128x64
      int r = c >> 3, off = (c & 7) << 3;
      g2l_b128(&As[r * 64 + off],
               A + (size_t)(mblk * 128 + r) * HID_ + kb * 64 + off);
    }
    for (int c = tid; c < 1024; c += 256) {      // Wt tile 128x64 (n-major)
      int r = c >> 3, off = (c & 7) << 3;
      g2l_b128(&Bs[r * 64 + off],
               Wt + (size_t)(nblk * 128 + r) * HID_ + kb * 64 + off);
    }
    async_join();
    __syncthreads();
#pragma unroll
    for (int ks = 0; ks < 2; ++ks) {
      v16bf a0 = load_frag(&As[(wr * 32 + 0) * 64 + ks * 32], 64);
      v16bf a1 = load_frag(&As[(wr * 32 + 16) * 64 + ks * 32], 64);
#pragma unroll
      for (int t = 0; t < 4; ++t) {
        v16bf b = load_frag(&Bs[(wc * 64 + t * 16) * 64 + ks * 32], 64);
        acc[0][t] = wmma_bf16(a0, b, acc[0][t]);
        acc[1][t] = wmma_bf16(a1, b, acc[1][t]);
      }
    }
    __syncthreads();
  }
#pragma unroll
  for (int i = 0; i < 2; ++i)
#pragma unroll
    for (int t = 0; t < 4; ++t) {
      int n = nblk * 128 + wc * 64 + t * 16 + (lane & 15);
      int m0 = mblk * 128 + wr * 32 + i * 16 + ((lane < 16) ? 0 : 8);
      if (MODE == 3) {
        int h = n >> 6, dd = n & 63;     // 8 consecutive m rows -> one b128
        *reinterpret_cast<uint4*>(
            out1 + ((size_t)(h * 64 + dd) * rowsM + m0)) = pack8_bf(acc[i][t]);
      } else if (MODE == 0) {
#pragma unroll
        for (int e = 0; e < 8; ++e)
          outF[(size_t)(m0 + e) * HID_ + n] = acc[i][t][e];
      } else if (MODE == 1) {
        int h = n >> 6, dd = n & 63;
#pragma unroll
        for (int e = 0; e < 8; ++e)
          out1[((size_t)h * rowsM + m0 + e) * 64 + dd] = f2bf(acc[i][t][e]);
      } else {
        int h = n >> 6, dd = n & 63;
        float b1 = bias1[n], b2 = bias2[n];
#pragma unroll
        for (int e = 0; e < 8; ++e) {
          out1[((size_t)h * rowsM + m0 + e) * 64 + dd] = f2bf(acc[i][t][e] + b1);
          out2[((size_t)h * rowsM + m0 + e) * 64 + dd] = f2bf(acc[i][t][e] + b2);
        }
      }
    }
}

// --------------------------------------- fused flash relative attention
// grid: B*H*(Q/64) blocks of 256 threads (8 waves). 64-row Q tile, 64-col K
// tiles, BD via 128-wide band GEMM scatter-added into S (rel_shift folded in).
__global__ __launch_bounds__(256)
void attention_kernel(const __bf16* __restrict__ qw,    // [H][B*Q][64]
                      const __bf16* __restrict__ qr,    // [H][B*Q][64]
                      const __bf16* __restrict__ kbuf,  // [H][B*K][64]
                      const __bf16* __restrict__ vbufT, // [H][64][B*K]
                      const __bf16* __restrict__ relb,  // [H][K][64]
                      const float* __restrict__ span_val,
                      __bf16* __restrict__ attn) {      // [B*Q][1024]
  __shared__ alignas(16) __bf16 qw_s[64 * 64];
  __shared__ alignas(16) __bf16 qr_s[64 * 64];
  __shared__ alignas(16) __bf16 kv_s[64 * 64];     // K tile, then V^T tile
  __shared__ alignas(16) __bf16 relp_s[128 * 64];  // rel band; later P tile
  __shared__ float S_s[64 * 64];
  __shared__ float m_s[64], lf_s[64], lm_s[64], al_s[64];

  int tid = threadIdx.x;
  int w = tid >> 5, lane = tid & 31;
  int i = w & 3, jh = w >> 2;             // row slab 16*i, col half jh

  int qblk = blockIdx.x & 31;
  int h = (blockIdx.x >> 5) & 15;
  int b = blockIdx.x >> 9;
  int q0 = qblk * 64;

  const size_t BQ = (size_t)B_ * Q_;
  const size_t BK = (size_t)B_ * K_;
  const __bf16* qwp = qw + ((size_t)h * BQ + (size_t)b * Q_ + q0) * 64;
  const __bf16* qrp = qr + ((size_t)h * BQ + (size_t)b * Q_ + q0) * 64;
  const __bf16* kp  = kbuf + ((size_t)h * BK + (size_t)b * K_) * 64;
  const __bf16* vtp = vbufT + (size_t)(h * 64) * BK + (size_t)b * K_;
  const __bf16* rp  = relb + (size_t)h * K_ * 64;
  float spanv = span_val[h];
  __bf16* p_s = relp_s;                    // P overlays dead rel band

  for (int c = tid; c < 512; c += 256) {   // stage q tiles once
    int r = c >> 3, off = (c & 7) << 3;
    g2l_b128(&qw_s[r * 64 + off], qwp + (size_t)r * 64 + off);
    g2l_b128(&qr_s[r * 64 + off], qrp + (size_t)r * 64 + off);
  }
  if (tid < 64) { m_s[tid] = -3e38f; lf_s[tid] = 0.f; lm_s[tid] = 0.f; }
  async_join();
  __syncthreads();

  // q fragments are loop-invariant: load once
  v16bf aw0 = load_frag(&qw_s[(i * 16) * 64 + 0], 64);
  v16bf aw1 = load_frag(&qw_s[(i * 16) * 64 + 32], 64);
  v16bf ar0 = load_frag(&qr_s[(i * 16) * 64 + 0], 64);
  v16bf ar1 = load_frag(&qr_s[(i * 16) * 64 + 32], 64);

  v8f o0 = (v8f){}, o1 = (v8f){};
  int ntiles = (M_ + q0 + 63) / 64 + 1;    // skip fully-causal-masked K tiles
  if (ntiles > K_ / 64) ntiles = K_ / 64;

  for (int kt = 0; kt < ntiles; ++kt) {
    int k0 = kt * 64;
    __syncthreads();                        // prior P@V reads complete
    for (int c = tid; c < 512; c += 256) {  // stage K tile
      int r = c >> 3, off = (c & 7) << 3;
      g2l_b128(&kv_s[r * 64 + off], kp + (size_t)(k0 + r) * 64 + off);
    }
    int jbase = k0 + (Q_ - 64) - q0;        // rel band origin (>= 0 always)
    for (int c = tid; c < 1024; c += 256) { // stage 128-row rel band
      int u = c >> 3, off = (c & 7) << 3;
      int j = jbase + u;
      if (j > K_ - 1) j = K_ - 1;           // OOB rows masked later anyway
      g2l_b128(&relp_s[u * 64 + off], rp + (size_t)j * 64 + off);
    }
    if (kt + 1 < ntiles) {                  // prefetch next K/V tiles
      int kn = k0 + 64;
      if (tid < 64)
        __builtin_prefetch(kp + (size_t)(kn + tid) * 64, 0, 3);
      else if (tid < 128)
        __builtin_prefetch(vtp + (size_t)(tid - 64) * BK + kn, 0, 3);
    }
    async_join();
    __syncthreads();

    // S = qw . k^T  (each wave: row slab i, col tiles 2*jh, 2*jh+1)
#pragma unroll
    for (int t = 0; t < 2; ++t) {
      int ct = jh * 2 + t;
      v8f acc = (v8f){};
      acc = wmma_bf16(aw0, load_frag(&kv_s[(ct * 16) * 64 + 0], 64), acc);
      acc = wmma_bf16(aw1, load_frag(&kv_s[(ct * 16) * 64 + 32], 64), acc);
#pragma unroll
      for (int e = 0; e < 8; ++e) {
        int m = i * 16 + e + ((lane < 16) ? 0 : 8);
        S_s[m * 64 + ct * 16 + (lane & 15)] = acc[e];
      }
    }
    __syncthreads();

    // E = qr . rel_band^T, scatter-add shifted: S[r][c] += E[r][c+63-r]
#pragma unroll
    for (int t = 0; t < 4; ++t) {
      int uc = jh * 64 + t * 16;
      v8f acc = (v8f){};
      acc = wmma_bf16(ar0, load_frag(&relp_s[uc * 64 + 0], 64), acc);
      acc = wmma_bf16(ar1, load_frag(&relp_s[uc * 64 + 32], 64), acc);
#pragma unroll
      for (int e = 0; e < 8; ++e) {
        int r = i * 16 + e + ((lane < 16) ? 0 : 8);
        int c = (uc + (lane & 15)) - 63 + r;  // unique (r,c) per producer
        if (c >= 0 && c < 64) atomicAdd(&S_s[r * 64 + c], acc[e]);
      }
    }
    __syncthreads();

    // threads 0-63: online softmax row pass; threads 64-255: stage V^T tile
    if (tid < 64) {
      int r = tid, qg = q0 + r;
      float mold = m_s[r];
      float mt = -3e38f;
#pragma unroll 8
      for (int c = 0; c < 64; ++c) {
        float v = S_s[r * 64 + c] * (1.0f / 64.0f);   // /head_size
        if (k0 + c > M_ + qg) v = -1e9f;              // causal
        mt = fmaxf(mt, v);
      }
      float mnew = fmaxf(mold, mt);
      float alpha = __expf(mold - mnew);
      float lf = 0.f, lm = 0.f;
#pragma unroll
      for (int c8 = 0; c8 < 8; ++c8) {
        struct P8 { __bf16 hh[8]; } pk;
#pragma unroll
        for (int e = 0; e < 8; ++e) {
          int c = c8 * 8 + e, kg = k0 + c;
          float v = S_s[r * 64 + c] * (1.0f / 64.0f);
          if (kg > M_ + qg) v = -1e9f;
          float p = __expf(v - mnew);
          float msk = 1.0f;
          if (kg >= M_) {                              // adaptive span ramp
            msk = ((float)(kg - M_) + spanv * (float)Q_) * (1.0f / 33.0f);
            msk = fminf(fmaxf(msk, 0.f), 1.f);
          }
          float pm = p * msk;
          lf += p; lm += pm;
          pk.hh[e] = f2bf(pm);
        }
        *reinterpret_cast<uint4*>(&p_s[r * 64 + c8 * 8]) =
            __builtin_bit_cast(uint4, pk);
      }
      m_s[r] = mnew;
      lf_s[r] = lf_s[r] * alpha + lf;
      lm_s[r] = lm_s[r] * alpha + lm;
      al_s[r] = alpha;
    } else {
      for (int c = tid - 64; c < 512; c += 192) {      // V^T rows d, cols kk
        int r = c >> 3, off = (c & 7) << 3;
        g2l_b128(&kv_s[r * 64 + off], vtp + (size_t)r * BK + k0 + off);
      }
    }
    async_join();
    __syncthreads();

    // O = alpha*O + P.V  (wave: row slab i, cols 32*jh .. +31)
#pragma unroll
    for (int e = 0; e < 8; ++e) {
      int row = i * 16 + e + ((lane < 16) ? 0 : 8);
      float a = al_s[row];
      o0[e] *= a; o1[e] *= a;
    }
    v16bf ap0 = load_frag(&p_s[(i * 16) * 64 + 0], 64);
    v16bf ap1 = load_frag(&p_s[(i * 16) * 64 + 32], 64);
    o0 = wmma_bf16(ap0, load_frag(&kv_s[(jh * 32 + 0) * 64 + 0], 64), o0);
    o0 = wmma_bf16(ap1, load_frag(&kv_s[(jh * 32 + 0) * 64 + 32], 64), o0);
    o1 = wmma_bf16(ap0, load_frag(&kv_s[(jh * 32 + 16) * 64 + 0], 64), o1);
    o1 = wmma_bf16(ap1, load_frag(&kv_s[(jh * 32 + 16) * 64 + 32], 64), o1);
  }

#pragma unroll
  for (int e = 0; e < 8; ++e) {
    int row = i * 16 + e + ((lane < 16) ? 0 : 8);
    float inv = 1.0f / (lm_s[row] + 1e-8f * lf_s[row]);  // mask renorm
    size_t base =
        ((size_t)b * Q_ + q0 + row) * HID_ + h * 64 + jh * 32 + (lane & 15);
    attn[base + 0]  = f2bf(o0[e] * inv);
    attn[base + 16] = f2bf(o1[e] * inv);
  }
}

// --------------------------------------------------------------- launcher
extern "C" void kernel_launch(void* const* d_in, const int* in_sizes, int n_in,
                              void* d_out, int out_size, void* d_ws,
                              size_t ws_size, hipStream_t stream) {
  (void)in_sizes; (void)n_in; (void)out_size; (void)ws_size;
  const float* query  = (const float*)d_in[0];
  const float* memory = (const float*)d_in[1];
  const float* Wq = (const float*)d_in[2];
  const float* Wk = (const float*)d_in[3];
  const float* Wv = (const float*)d_in[4];
  const float* Wo = (const float*)d_in[5];
  const float* Wr = (const float*)d_in[6];
  const float* gamma_mem = (const float*)d_in[7];
  const float* beta_mem  = (const float*)d_in[8];
  const float* gamma_q   = (const float*)d_in[9];
  const float* beta_q    = (const float*)d_in[10];
  const float* r_w_bias  = (const float*)d_in[11];
  const float* r_r_bias  = (const float*)d_in[12];
  const float* span_val  = (const float*)d_in[13];

  char* ws = (char*)d_ws;
  size_t off = 0;
  auto alloc = [&](size_t bytes) -> void* {
    void* p = ws + off;
    off = (off + bytes + 255) & ~(size_t)255;
    return p;
  };
  const size_t BQ = (size_t)B_ * Q_, BK = (size_t)B_ * K_;
  __bf16* WqT = (__bf16*)alloc((size_t)HID_ * HID_ * 2);
  __bf16* WkT = (__bf16*)alloc((size_t)HID_ * HID_ * 2);
  __bf16* WvT = (__bf16*)alloc((size_t)HID_ * HID_ * 2);
  __bf16* WrT = (__bf16*)alloc((size_t)HID_ * HID_ * 2);
  __bf16* WoT = (__bf16*)alloc((size_t)HID_ * HID_ * 2);
  __bf16* kvln = (__bf16*)alloc(BK * HID_ * 2);
  __bf16* qln  = (__bf16*)alloc(BQ * HID_ * 2);
  __bf16* posb = (__bf16*)alloc((size_t)K_ * HID_ * 2);
  __bf16* qwb  = (__bf16*)alloc((size_t)H_ * BQ * 64 * 2);
  __bf16* qrb  = (__bf16*)alloc((size_t)H_ * BQ * 64 * 2);
  __bf16* kbuf = (__bf16*)alloc((size_t)H_ * BK * 64 * 2);
  __bf16* vbufT = (__bf16*)alloc((size_t)H_ * 64 * BK * 2);
  __bf16* relb = (__bf16*)alloc((size_t)H_ * K_ * 64 * 2);
  __bf16* attn = (__bf16*)alloc(BQ * HID_ * 2);

  dim3 tg(32, 32);
  cvt_transpose_kernel<<<tg, 256, 0, stream>>>(Wq, WqT);
  cvt_transpose_kernel<<<tg, 256, 0, stream>>>(Wk, WkT);
  cvt_transpose_kernel<<<tg, 256, 0, stream>>>(Wv, WvT);
  cvt_transpose_kernel<<<tg, 256, 0, stream>>>(Wr, WrT);
  cvt_transpose_kernel<<<tg, 256, 0, stream>>>(Wo, WoT);

  layernorm_kernel<<<B_ * K_, 256, 0, stream>>>(query, memory, gamma_mem,
                                                beta_mem, kvln, 1);
  layernorm_kernel<<<B_ * Q_, 256, 0, stream>>>(query, nullptr, gamma_q,
                                                beta_q, qln, 0);
  posemb_kernel<<<(K_ * HID_) / 256, 256, 0, stream>>>(posb);

  dim3 gq((int)(BQ / 128), HID_ / 128);
  dim3 gk((int)(BK / 128), HID_ / 128);
  dim3 gr(K_ / 128, HID_ / 128);
  gemm_bf16_kernel<2><<<gq, 256, 0, stream>>>(qln, WqT, (int)BQ, nullptr, qwb,
                                              qrb, r_w_bias, r_r_bias);
  gemm_bf16_kernel<1><<<gk, 256, 0, stream>>>(kvln, WkT, (int)BK, nullptr,
                                              kbuf, nullptr, nullptr, nullptr);
  gemm_bf16_kernel<3><<<gk, 256, 0, stream>>>(kvln, WvT, (int)BK, nullptr,
                                              vbufT, nullptr, nullptr, nullptr);
  gemm_bf16_kernel<1><<<gr, 256, 0, stream>>>(posb, WrT, K_, nullptr, relb,
                                              nullptr, nullptr, nullptr);

  attention_kernel<<<B_ * H_ * (Q_ / 64), 256, 0, stream>>>(
      qwb, qrb, kbuf, vbufT, relb, span_val, attn);

  gemm_bf16_kernel<0><<<gq, 256, 0, stream>>>(attn, WoT, (int)BQ,
                                              (float*)d_out, nullptr, nullptr,
                                              nullptr, nullptr);
}
